// ProxyNCALayer_18451179503909
// MI455X (gfx1250) — matrix operand order repaired and checked by reading, not compiled, not run on hardware
//
#include <hip/hip_runtime.h>

typedef __attribute__((ext_vector_type(8)))  float  v8f;
typedef __attribute__((ext_vector_type(16))) __bf16 v16bf;
typedef __attribute__((ext_vector_type(8)))  __bf16 v8bf;
typedef __attribute__((ext_vector_type(4)))  __bf16 v4bf;

#define B_ROWS 4096
#define C_ROWS 10000
#define DDIM   512
#define EPSN   1e-12f
#define SCALE  3.0f

#define TM   128
#define TN   128
#define KBLK 64
#define LDB  72   // bf16 row stride in LDS panels (144 B, 16-B aligned)

// ---------------- Phase 1: per-row inverse norms ----------------
__global__ __launch_bounds__(256)
void rownorm_kernel(const float* __restrict__ x, const float* __restrict__ p,
                    float* __restrict__ invx, float* __restrict__ xsq,
                    float* __restrict__ invp, float* __restrict__ psq) {
    const int wid  = threadIdx.x >> 5;
    const int lane = threadIdx.x & 31;
    const int row  = blockIdx.x * 8 + wid;

    const float* src;
    float* invDst;
    float* sqDst;
    if (row < B_ROWS) {
        src = x + (size_t)row * DDIM; invDst = invx + row; sqDst = xsq + row;
    } else {
        const int r = row - B_ROWS;
        if (r >= C_ROWS) return;
        src = p + (size_t)r * DDIM; invDst = invp + r; sqDst = psq + r;
    }

    float s = 0.0f;
    #pragma unroll
    for (int i = 0; i < DDIM / 32; ++i) {
        const float v = src[lane + i * 32];
        s = fmaf(v, v, s);
    }
    #pragma unroll
    for (int off = 16; off > 0; off >>= 1) s += __shfl_xor(s, off, 32);

    const float m    = fmaxf(s, EPSN);
    const float rinv = rsqrtf(m);
    if (lane == 0) {
        *invDst = SCALE * rinv;
        *sqDst  = (SCALE * SCALE) * s * rinv * rinv;
    }
}

// ---------------- Phase 1b: one-time normalize + bf16 hi/lo split ----------------
__global__ __launch_bounds__(256)
void split_kernel(const float* __restrict__ x, const float* __restrict__ p,
                  const float* __restrict__ invx, const float* __restrict__ invp,
                  __bf16* __restrict__ xhi, __bf16* __restrict__ xlo,
                  __bf16* __restrict__ phi, __bf16* __restrict__ plo) {
    const size_t XQ = (size_t)B_ROWS * (DDIM / 4);
    const size_t PQ = (size_t)C_ROWS * (DDIM / 4);
    size_t idx = (size_t)blockIdx.x * 256 + threadIdx.x;  // float4 index

    const float* src;
    const float* inv;
    __bf16 *dhi, *dlo;
    size_t row, c4;
    if (idx < XQ) {
        row = idx >> 7; c4 = (idx & 127) << 2;
        src = x; inv = invx; dhi = xhi; dlo = xlo;
    } else {
        idx -= XQ;
        if (idx >= PQ) return;
        row = idx >> 7; c4 = (idx & 127) << 2;
        src = p; inv = invp; dhi = phi; dlo = plo;
    }

    const float4 v = *(const float4*)(src + row * DDIM + c4);
    const float  s = inv[row];
    const float f0 = v.x * s, f1 = v.y * s, f2 = v.z * s, f3 = v.w * s;
    const __bf16 h0 = (__bf16)f0, h1 = (__bf16)f1, h2 = (__bf16)f2, h3 = (__bf16)f3;
    const v4bf hv = {h0, h1, h2, h3};
    const v4bf lv = {(__bf16)(f0 - (float)h0), (__bf16)(f1 - (float)h1),
                     (__bf16)(f2 - (float)h2), (__bf16)(f3 - (float)h3)};
    *(v4bf*)(dhi + row * DDIM + c4) = hv;
    *(v4bf*)(dlo + row * DDIM + c4) = lv;
}

// -------- CDNA5 async Global -> LDS copy (ASYNCcnt path) --------
__device__ __forceinline__ void async_copy_b128(const void* gptr, unsigned ldsByteAddr) {
    asm volatile("global_load_async_to_lds_b128 %0, %1, off"
                 :: "v"(ldsByteAddr), "v"(gptr) : "memory");
}
__device__ __forceinline__ void wait_async0() {
#if __has_builtin(__builtin_amdgcn_s_wait_asynccnt)
    __builtin_amdgcn_s_wait_asynccnt(0);
#else
    asm volatile("s_wait_asynccnt 0x0" ::: "memory");
#endif
}

// -------- bf16 fragment loads (ISA 7.12.2 16-bit layouts) --------
__device__ __forceinline__ v16bf ld_frag_a(const __bf16* p) {
    const v8bf c0 = *(const v8bf*)(p);
    const v8bf c1 = *(const v8bf*)(p + 16);
    return __builtin_shufflevector(c0, c1, 0, 1, 2, 3, 4, 5, 6, 7,
                                           8, 9, 10, 11, 12, 13, 14, 15);
}
__device__ __forceinline__ v16bf ld_frag_b(const __bf16* p) {
    const v8bf c0 = *(const v8bf*)(p);
    const v8bf c1 = *(const v8bf*)(p + 16);
    return __builtin_shufflevector(c0, c1, 0, 1, 2, 3, 4, 5, 6, 7,
                                           8, 9, 10, 11, 12, 13, 14, 15);
}

#define WMMA_BF16(A, B, C) \
    __builtin_amdgcn_wmma_f32_16x16x32_bf16(false, (A), false, (B), (short)0, (C), false, false)

// ---------------- Phase 2 (fast): double-buffered async-staged split-bf16 GEMM ----------------
__global__ __launch_bounds__(256)
void proxynca_wmma_async(const __bf16* __restrict__ xhi, const __bf16* __restrict__ xlo,
                         const __bf16* __restrict__ phi, const __bf16* __restrict__ plo,
                         const float* __restrict__ xsq, const float* __restrict__ psq,
                         float* __restrict__ out) {
    __shared__ __bf16 Ah[2][TM * LDB];
    __shared__ __bf16 Al[2][TM * LDB];
    __shared__ __bf16 Bh[2][TN * LDB];
    __shared__ __bf16 Bl[2][TN * LDB];

    const int tid  = threadIdx.x;
    const int lane = tid & 31;
    const int wid  = tid >> 5;
    const int wm   = wid & 3;
    const int wn   = wid >> 2;

    const int ntiles = (C_ROWS + TN - 1) / TN;
    const int tileM  = (blockIdx.x / ntiles) * TM;
    const int tileN  = (blockIdx.x % ntiles) * TN;

    const int fr     = lane & 15;
    const int base_a = (lane >> 4) << 3;
    const int base_b = (lane >> 4) << 3;

    v8f acc[2][4] = {};

    // Issue all async b128 copies for one K-block into buffer `buf` (no wait).
    auto stage = [&](int buf, int k0) {
        #pragma unroll
        for (int pan = 0; pan < 4; ++pan) {
            const __bf16* gbase;
            __bf16* lbase;
            int rowbase;
            if      (pan == 0) { gbase = xhi; lbase = Ah[buf]; rowbase = tileM; }
            else if (pan == 1) { gbase = xlo; lbase = Al[buf]; rowbase = tileM; }
            else if (pan == 2) { gbase = phi; lbase = Bh[buf]; rowbase = tileN; }
            else               { gbase = plo; lbase = Bl[buf]; rowbase = tileN; }
            #pragma unroll
            for (int ii = 0; ii < 4; ++ii) {
                const int li = tid + ii * 256;      // 0..1023
                const int r  = li >> 3;             // 0..127
                const int cB = (li & 7) << 4;       // byte offset 0..112
                int grow = rowbase + r;
                if (pan >= 2 && grow >= C_ROWS) grow = C_ROWS - 1;  // clamp ragged tile
                const char* gp  = (const char*)(gbase + (size_t)grow * DDIM + k0) + cB;
                const unsigned l = (unsigned)(unsigned long long)
                                   ((const char*)(lbase + r * LDB) + cB);
                async_copy_b128(gp, l);
            }
        }
    };

    // Prologue: fill buffer 0.
    stage(0, 0);
    wait_async0();
    __syncthreads();

    int cur = 0;
    for (int k0 = 0; k0 < DDIM; k0 += KBLK) {
        const int nxt = cur ^ 1;
        if (k0 + KBLK < DDIM) stage(nxt, k0 + KBLK);  // DMA next block behind compute

        #pragma unroll
        for (int kk = 0; kk < KBLK; kk += 32) {
            v16bf ah[2], al[2], bh[4], bl[4];
            #pragma unroll
            for (int i = 0; i < 2; ++i) {
                const int off = (wm * 32 + i * 16 + fr) * LDB + kk + base_a;
                ah[i] = ld_frag_a(&Ah[cur][off]);
                al[i] = ld_frag_a(&Al[cur][off]);
            }
            #pragma unroll
            for (int j = 0; j < 4; ++j) {
                const int off = (wn * 64 + j * 16 + fr) * LDB + kk + base_b;
                bh[j] = ld_frag_b(&Bh[cur][off]);
                bl[j] = ld_frag_b(&Bl[cur][off]);
            }
            #pragma unroll
            for (int i = 0; i < 2; ++i)
                #pragma unroll
                for (int j = 0; j < 4; ++j) {
                    acc[i][j] = WMMA_BF16(ah[i], bh[j], acc[i][j]);
                    acc[i][j] = WMMA_BF16(ah[i], bl[j], acc[i][j]);
                    acc[i][j] = WMMA_BF16(al[i], bh[j], acc[i][j]);
                }
        }

        wait_async0();      // own prefetch landed
        __syncthreads();    // everyone finished reading `cur` and prefetching `nxt`
        cur = nxt;
    }

    const int mh = (lane >> 4) << 3;
    const int n  = lane & 15;
    #pragma unroll
    for (int i = 0; i < 2; ++i) {
        #pragma unroll
        for (int j = 0; j < 4; ++j) {
            const int colBase = tileN + wn * 64 + j * 16;
            if (colBase >= C_ROWS) continue;
            const int col = colBase + n;
            const float pq = psq[col];
            #pragma unroll
            for (int v = 0; v < 8; ++v) {
                const int row = tileM + wm * 32 + i * 16 + mh + v;
                out[(size_t)row * C_ROWS + col] = xsq[row] + pq - 2.0f * acc[i][j][v];
            }
        }
    }
}

// ---------------- Phase 2 (fallback): in-kernel split, small-ws path ----------------
__global__ __launch_bounds__(256)
void proxynca_wmma_fallback(const float* __restrict__ x, const float* __restrict__ p,
                            const float* __restrict__ invx, const float* __restrict__ xsq,
                            const float* __restrict__ invp, const float* __restrict__ psq,
                            float* __restrict__ out) {
    __shared__ __bf16 Ah[TM * LDB];
    __shared__ __bf16 Al[TM * LDB];
    __shared__ __bf16 Bh[TN * LDB];
    __shared__ __bf16 Bl[TN * LDB];

    const int tid  = threadIdx.x;
    const int lane = tid & 31;
    const int wid  = tid >> 5;
    const int wm   = wid & 3;
    const int wn   = wid >> 2;

    const int ntiles = (C_ROWS + TN - 1) / TN;
    const int tileM  = (blockIdx.x / ntiles) * TM;
    const int tileN  = (blockIdx.x % ntiles) * TN;

    const int fr     = lane & 15;
    const int base_a = (lane >> 4) << 3;
    const int base_b = (lane >> 4) << 3;

    v8f acc[2][4] = {};

    for (int k0 = 0; k0 < DDIM; k0 += KBLK) {
        __syncthreads();
        #pragma unroll
        for (int i = 0; i < 8; ++i) {
            const int li = tid + i * 256;
            const int r  = li >> 4;
            const int c4 = (li & 15) << 2;
            {
                const int   grow = tileM + r;
                const float4 v   = *(const float4*)(x + (size_t)grow * DDIM + k0 + c4);
                const float  s   = invx[grow];
                const float f0 = v.x * s, f1 = v.y * s, f2 = v.z * s, f3 = v.w * s;
                const __bf16 h0 = (__bf16)f0, h1 = (__bf16)f1, h2 = (__bf16)f2, h3 = (__bf16)f3;
                const v4bf hv = {h0, h1, h2, h3};
                const v4bf lv = {(__bf16)(f0 - (float)h0), (__bf16)(f1 - (float)h1),
                                 (__bf16)(f2 - (float)h2), (__bf16)(f3 - (float)h3)};
                *(v4bf*)(&Ah[r * LDB + c4]) = hv;
                *(v4bf*)(&Al[r * LDB + c4]) = lv;
            }
            {
                const int grow = tileN + r;
                float f0 = 0.f, f1 = 0.f, f2 = 0.f, f3 = 0.f;
                if (grow < C_ROWS) {
                    const float4 v = *(const float4*)(p + (size_t)grow * DDIM + k0 + c4);
                    const float  s = invp[grow];
                    f0 = v.x * s; f1 = v.y * s; f2 = v.z * s; f3 = v.w * s;
                }
                const __bf16 h0 = (__bf16)f0, h1 = (__bf16)f1, h2 = (__bf16)f2, h3 = (__bf16)f3;
                const v4bf hv = {h0, h1, h2, h3};
                const v4bf lv = {(__bf16)(f0 - (float)h0), (__bf16)(f1 - (float)h1),
                                 (__bf16)(f2 - (float)h2), (__bf16)(f3 - (float)h3)};
                *(v4bf*)(&Bh[r * LDB + c4]) = hv;
                *(v4bf*)(&Bl[r * LDB + c4]) = lv;
            }
        }
        __syncthreads();

        #pragma unroll
        for (int kk = 0; kk < KBLK; kk += 32) {
            v16bf ah[2], al[2], bh[4], bl[4];
            #pragma unroll
            for (int i = 0; i < 2; ++i) {
                const int off = (wm * 32 + i * 16 + fr) * LDB + kk + base_a;
                ah[i] = ld_frag_a(&Ah[off]);
                al[i] = ld_frag_a(&Al[off]);
            }
            #pragma unroll
            for (int j = 0; j < 4; ++j) {
                const int off = (wn * 64 + j * 16 + fr) * LDB + kk + base_b;
                bh[j] = ld_frag_b(&Bh[off]);
                bl[j] = ld_frag_b(&Bl[off]);
            }
            #pragma unroll
            for (int i = 0; i < 2; ++i)
                #pragma unroll
                for (int j = 0; j < 4; ++j) {
                    acc[i][j] = WMMA_BF16(ah[i], bh[j], acc[i][j]);
                    acc[i][j] = WMMA_BF16(ah[i], bl[j], acc[i][j]);
                    acc[i][j] = WMMA_BF16(al[i], bh[j], acc[i][j]);
                }
        }
    }

    const int mh = (lane >> 4) << 3;
    const int n  = lane & 15;
    #pragma unroll
    for (int i = 0; i < 2; ++i) {
        #pragma unroll
        for (int j = 0; j < 4; ++j) {
            const int colBase = tileN + wn * 64 + j * 16;
            if (colBase >= C_ROWS) continue;
            const int col = colBase + n;
            const float pq = psq[col];
            #pragma unroll
            for (int v = 0; v < 8; ++v) {
                const int row = tileM + wm * 32 + i * 16 + mh + v;
                out[(size_t)row * C_ROWS + col] = xsq[row] + pq - 2.0f * acc[i][j][v];
            }
        }
    }
}

extern "C" void kernel_launch(void* const* d_in, const int* in_sizes, int n_in,
                              void* d_out, int out_size, void* d_ws, size_t ws_size,
                              hipStream_t stream) {
    const float* x = (const float*)d_in[0];
    const float* p = (const float*)d_in[1];
    float* out = (float*)d_out;

    // ws layout: [invx|xsq|invp|psq] (f32) then bf16 split matrices
    float* ws   = (float*)d_ws;
    float* invx = ws;
    float* xsq  = ws + B_ROWS;
    float* invp = ws + 2 * B_ROWS;
    float* psq  = ws + 2 * B_ROWS + C_ROWS;
    const size_t normBytes  = (size_t)(2 * B_ROWS + 2 * C_ROWS) * sizeof(float);  // 112,768
    const size_t xElems     = (size_t)B_ROWS * DDIM;
    const size_t pElems     = (size_t)C_ROWS * DDIM;
    const size_t splitBytes = (xElems + pElems) * 2 /*hi+lo*/ * sizeof(__bf16);

    const int rows    = B_ROWS + C_ROWS;
    const int blocks1 = (rows + 7) / 8;
    rownorm_kernel<<<blocks1, 256, 0, stream>>>(x, p, invx, xsq, invp, psq);

    const int ntiles = (C_ROWS + TN - 1) / TN;  // 79
    const int mtiles = B_ROWS / TM;             // 32

    if (ws_size >= normBytes + splitBytes) {
        __bf16* xhi = (__bf16*)((char*)d_ws + normBytes);
        __bf16* xlo = xhi + xElems;
        __bf16* phi = xlo + xElems;
        __bf16* plo = phi + pElems;

        const size_t q       = (xElems + pElems) / 4;  // float4 count (divisible)
        const int    blocks2 = (int)((q + 255) / 256);
        split_kernel<<<blocks2, 256, 0, stream>>>(x, p, invx, invp, xhi, xlo, phi, plo);

        proxynca_wmma_async<<<mtiles * ntiles, 256, 0, stream>>>(
            xhi, xlo, phi, plo, xsq, psq, out);
    } else {
        proxynca_wmma_fallback<<<mtiles * ntiles, 256, 0, stream>>>(
            x, p, invx, xsq, invp, psq, out);
    }
}
